// EfficientBioNeuron_86981677678632
// MI455X (gfx1250) — matrix operand order repaired and checked by reading, not compiled, not run on hardware
//
#include <hip/hip_runtime.h>
#include <hip/hip_bf16.h>

// ---------------------------------------------------------------------------
// Problem constants (match reference)
// ---------------------------------------------------------------------------
#define BB 4
#define TT 1024
#define DD 1024
#define NHH 16
#define DHH 64
#define SOMA_HH 3072

typedef __attribute__((ext_vector_type(16))) __bf16 v16bf;
typedef __attribute__((ext_vector_type(8)))  float  v8f;
typedef __attribute__((ext_vector_type(4)))  int    v4i;

// Optional CDNA5 async global->LDS path (ASYNCcnt-tracked DMA).
#if defined(__has_builtin)
#  if __has_builtin(__builtin_amdgcn_global_load_async_to_lds_b128)
#    define HAVE_ASYNC_LDS 1
#  endif
#endif
#ifndef HAVE_ASYNC_LDS
#  define HAVE_ASYNC_LDS 0
#endif

#if HAVE_ASYNC_LDS
typedef __attribute__((address_space(1))) v4i gv4i;   // global 128b chunk
typedef __attribute__((address_space(3))) v4i lv4i;   // LDS 128b chunk
__device__ __forceinline__ void async_copy_b128(const float* gsrc, float* ldst) {
    __builtin_amdgcn_global_load_async_to_lds_b128((gv4i*)gsrc, (lv4i*)ldst, 0, 0);
}
#endif

// Native conversion: backend emits v_cvt_pk_bf16_f32.
__device__ __forceinline__ __bf16 f2bf(float f) { return (__bf16)f; }

// Pack two floats into one dword of bf16 (lo = a, hi = b).
__device__ __forceinline__ unsigned pack_bf16_2(float a, float b) {
    unsigned short lo = __builtin_bit_cast(unsigned short, (__bf16)a);
    unsigned short hi = __builtin_bit_cast(unsigned short, (__bf16)b);
    return ((unsigned)hi << 16) | (unsigned)lo;
}

__device__ __forceinline__ float gelu_f(float x) {
    return 0.5f * x * (1.0f + erff(x * 0.70710678118654752f));
}

// A-matrix (16x32 bf16) element->K mapping per ISA 7.12.2:
// lanes 0-15: VGPR0..3 = K 0..7, VGPR4..7 = K 16..23
// lanes 16-31: VGPR0..3 = K 8..15, VGPR4..7 = K 24..31
__device__ __forceinline__ int a_k(int lane, int e) {
    int v = e >> 1, s = e & 1;
    int half8 = (lane >> 4) << 3;
    return (v < 4) ? (half8 + v * 2 + s) : (16 + half8 + (v - 4) * 2 + s);
}

__device__ __forceinline__ v8f wmma_bf16(v16bf a, v16bf b, v8f c) {
    return __builtin_amdgcn_wmma_f32_16x16x32_bf16(false, a, false, b, (short)0, c, false, false);
}

// ---------------------------------------------------------------------------
// bf16-WMMA GEMM:  C[M,N] = epi(A[M,K] @ W[K,N])
// EPI: 0 = none, 1 = +bias then exact GeLU, 2 = +bias +residual
// Block: 256 threads = 8 waves; 64x128 block tile; wave -> 32x32 subtile
// (2 A-frags x 2 B-frags -> 4 WMMAs / K-step, 2x fragment reuse).
// W tile is DMA'd to LDS via global_load_async_to_lds_b128 (ASYNCcnt),
// A tile is a packed-bf16 cooperative fill.
// Requires: M % 64 == 0, N % 128 == 0, K % 32 == 0 (true for all uses).
// ---------------------------------------------------------------------------
template <int EPI>
__global__ __launch_bounds__(256) void gemm_bf16_kernel(
    const float* __restrict__ A, const float* __restrict__ W,
    const float* __restrict__ bias, const float* __restrict__ res,
    float* __restrict__ C, int M, int N, int K)
{
    __shared__ __align__(16) __bf16 As[64][40];     // 32 cols + pad (16B mult)
    __shared__ __align__(16) float  Wf32[32][132];  // 128 cols + pad (16B mult)

    const int tid  = threadIdx.x;
    const int lane = tid & 31;
    const int wave = tid >> 5;
    const int wm = wave & 1, wn = wave >> 1;       // 2x4 wave grid
    const int row0 = blockIdx.y * 64;
    const int col0 = blockIdx.x * 128;
    const int half = lane >> 4, nn = lane & 15;

    v8f c00 = {}, c01 = {}, c10 = {}, c11 = {};

    for (int k0 = 0; k0 < K; k0 += 32) {
        // ---- W tile (32x128 fp32): async DMA to LDS, 4 x b128 per thread
#if HAVE_ASYNC_LDS
        #pragma unroll
        for (int i = 0; i < 4; ++i) {
            const int idx = tid + i * 256;          // 1024 float4 chunks
            const int r = idx >> 5, q = idx & 31;
            async_copy_b128(&W[(size_t)(k0 + r) * N + col0 + 4 * q],
                            &Wf32[r][4 * q]);
        }
#else
        #pragma unroll
        for (int i = 0; i < 4; ++i) {
            const int idx = tid + i * 256;
            const int r = idx >> 5, q = idx & 31;
            *(float4*)&Wf32[r][4 * q] =
                *(const float4*)&W[(size_t)(k0 + r) * N + col0 + 4 * q];
        }
#endif
        // ---- A tile (64x32): 8 elements/thread, b128 load + packed bf16 store
        {
            const int r = tid >> 2, q = tid & 3;
            const float* src = &A[(size_t)(row0 + r) * K + k0 + 8 * q];
            float4 f0 = *(const float4*)(src);
            float4 f1 = *(const float4*)(src + 4);
            uint4 p;
            p.x = pack_bf16_2(f0.x, f0.y); p.y = pack_bf16_2(f0.z, f0.w);
            p.z = pack_bf16_2(f1.x, f1.y); p.w = pack_bf16_2(f1.z, f1.w);
            *(uint4*)&As[r][8 * q] = p;
        }
        if (k0 + 32 < K) {  // speculative prefetch of next A tile
            __builtin_prefetch(&A[(size_t)(row0 + (tid >> 2)) * K + (k0 + 32)], 0, 1);
        }
#if HAVE_ASYNC_LDS
        asm volatile("s_wait_asynccnt 0" ::: "memory");
#endif
        __syncthreads();

        // ---- fragments
        v16bf a0, a1, b0, b1;
        const int m0 = wm * 32 + nn;
        #pragma unroll
        for (int e = 0; e < 16; ++e) {
            const int k = half * 16 + e;
            a0[e] = As[m0][k];
            a1[e] = As[m0 + 16][k];
        }
        const int kb = half * 16;
        #pragma unroll
        for (int e = 0; e < 16; ++e) {
            b0[e] = f2bf(Wf32[kb + e][wn * 32 + nn]);
            b1[e] = f2bf(Wf32[kb + e][wn * 32 + 16 + nn]);
        }
        c00 = wmma_bf16(a0, b0, c00);
        c01 = wmma_bf16(a0, b1, c01);
        c10 = wmma_bf16(a1, b0, c10);
        c11 = wmma_bf16(a1, b1, c11);
        __syncthreads();
    }

    // ---- epilogue: C/D layout row = r + 8*half (+16 per M-subtile)
    #pragma unroll
    for (int r = 0; r < 8; ++r) {
        #pragma unroll
        for (int mf = 0; mf < 2; ++mf) {
            const int m = row0 + wm * 32 + mf * 16 + half * 8 + r;
            #pragma unroll
            for (int nf = 0; nf < 2; ++nf) {
                const int n = col0 + wn * 32 + nf * 16 + nn;
                float v;
                if (mf == 0) v = (nf == 0) ? c00[r] : c01[r];
                else         v = (nf == 0) ? c10[r] : c11[r];
                if constexpr (EPI == 1) { v = gelu_f(v + bias[n]); }
                else if constexpr (EPI == 2) { v = v + bias[n] + res[(size_t)m * N + n]; }
                C[(size_t)m * N + n] = v;
            }
        }
    }
}

// ---------------------------------------------------------------------------
// Flash attention with WMMA. One wave per (b, h, 16-query tile).
// qkv layout: (B*T, 3*D) rows; Q cols [0,D), K [D,2D), V [2D,3D), col=h*64+dh.
// V gets +nmda[b,h,dh]*sigmoid(decay[h,dh]) folded in.
// Output: branch (B,T,D) with d = h*64+dh.
// ---------------------------------------------------------------------------
__global__ __launch_bounds__(256) void attn_kernel(
    const float* __restrict__ qkv, const float* __restrict__ nmda,
    const float* __restrict__ decay, float* __restrict__ branch)
{
    __shared__ __bf16 pbuf[8][16 * 32];   // per-wave P scratch

    const int lane = threadIdx.x & 31;
    const int wave = threadIdx.x >> 5;
    const int task = blockIdx.x * 8 + wave;       // B * NH * (T/16) = 4096
    const int qt = task & 63;
    const int h  = (task >> 6) & 15;
    const int b  = task >> 10;
    const int q0 = qt * 16;
    const int half = lane >> 4, nn = lane & 15;

    const size_t LD = 3 * DD;
    const float* qbase = qkv + (size_t)b * TT * LD + h * DHH;
    const float* kbase = qbase + DD;
    const float* vbase = qbase + 2 * DD;

    // Q A-fragments (16 x 64 = two 16x32 chunks)
    v16bf aq0, aq1;
    {
        const float* qrow = qbase + (size_t)(q0 + nn) * LD;
        #pragma unroll
        for (int e = 0; e < 16; ++e) {
            const int k = half * 16 + e;
            aq0[e] = f2bf(qrow[k]);
            aq1[e] = f2bf(qrow[32 + k]);
        }
    }
    // per dh-tile V bias
    float adj[4];
    #pragma unroll
    for (int t = 0; t < 4; ++t) {
        const int dh = t * 16 + nn;
        const float dcy = decay[h * DHH + dh];
        adj[t] = nmda[((size_t)b * NHH + h) * DHH + dh] / (1.0f + __expf(-dcy));
    }

    float Mrow[8], Lrow[8];
    v8f o0 = {}, o1 = {}, o2 = {}, o3 = {};
    #pragma unroll
    for (int r = 0; r < 8; ++r) { Mrow[r] = -1e30f; Lrow[r] = 0.0f; }

    const int nkb = ((q0 + 15) >> 5) + 1;    // 32-key blocks, causal
    const float scale = 0.125f;              // 1/sqrt(64)

    for (int kbi = 0; kbi < nkb; ++kbi) {
        const int kb = kbi * 32;

        // K^T B-fragments: N = key, gemmK = dh; rows contiguous -> b128 loads
        v16bf bk00, bk01, bk10, bk11;
        {
            const float* kr0 = kbase + (size_t)(kb + nn) * LD + half * 16;
            const float* kr1 = kbase + (size_t)(kb + 16 + nn) * LD + half * 16;
            #pragma unroll
            for (int v = 0; v < 4; ++v) {
                float4 c0 = *(const float4*)(kr0 + 4 * v);
                float4 c1 = *(const float4*)(kr0 + 32 + 4 * v);
                float4 d0 = *(const float4*)(kr1 + 4 * v);
                float4 d1 = *(const float4*)(kr1 + 32 + 4 * v);
                bk00[4*v+0] = f2bf(c0.x); bk00[4*v+1] = f2bf(c0.y);
                bk00[4*v+2] = f2bf(c0.z); bk00[4*v+3] = f2bf(c0.w);
                bk01[4*v+0] = f2bf(c1.x); bk01[4*v+1] = f2bf(c1.y);
                bk01[4*v+2] = f2bf(c1.z); bk01[4*v+3] = f2bf(c1.w);
                bk10[4*v+0] = f2bf(d0.x); bk10[4*v+1] = f2bf(d0.y);
                bk10[4*v+2] = f2bf(d0.z); bk10[4*v+3] = f2bf(d0.w);
                bk11[4*v+0] = f2bf(d1.x); bk11[4*v+1] = f2bf(d1.y);
                bk11[4*v+2] = f2bf(d1.z); bk11[4*v+3] = f2bf(d1.w);
            }
        }
        v8f s0 = {}, s1 = {};
        s0 = wmma_bf16(aq0, bk00, s0); s0 = wmma_bf16(aq1, bk01, s0);
        s1 = wmma_bf16(aq0, bk10, s1); s1 = wmma_bf16(aq1, bk11, s1);

        const bool need_mask = (kb + 31 > q0);
        #pragma unroll
        for (int r = 0; r < 8; ++r) {
            const int q = q0 + half * 8 + r;
            float v0 = s0[r] * scale;
            float v1 = s1[r] * scale;
            if (need_mask) {
                if (kb + nn > q)      v0 = -1e30f;
                if (kb + 16 + nn > q) v1 = -1e30f;
            }
            float mx = fmaxf(v0, v1);
            #pragma unroll
            for (int off = 8; off >= 1; off >>= 1) mx = fmaxf(mx, __shfl_xor(mx, off));
            const float mnew = fmaxf(Mrow[r], mx);
            const float p0 = __expf(v0 - mnew);
            const float p1 = __expf(v1 - mnew);
            const float fac = __expf(Mrow[r] - mnew);
            float ps = p0 + p1;
            #pragma unroll
            for (int off = 8; off >= 1; off >>= 1) ps += __shfl_xor(ps, off);
            Lrow[r] = Lrow[r] * fac + ps;
            Mrow[r] = mnew;
            o0[r] *= fac; o1[r] *= fac; o2[r] *= fac; o3[r] *= fac;
            pbuf[wave][(half * 8 + r) * 32 + nn]      = f2bf(p0);
            pbuf[wave][(half * 8 + r) * 32 + 16 + nn] = f2bf(p1);
        }
        asm volatile("s_wait_dscnt 0" ::: "memory");

        // P as A-fragment (16x32), gemmK = key-in-block
        v16bf ap;
        #pragma unroll
        for (int e = 0; e < 16; ++e) ap[e] = pbuf[wave][nn * 32 + half * 16 + e];

        // V B-fragments: N = dh-tile, gemmK = key
        v16bf bv0, bv1, bv2, bv3;
        {
            const float* vr = vbase + (size_t)(kb + half * 16) * LD;
            #pragma unroll
            for (int e = 0; e < 16; ++e) {
                const size_t o = (size_t)e * LD;
                bv0[e] = f2bf(vr[o + 0  + nn] + adj[0]);
                bv1[e] = f2bf(vr[o + 16 + nn] + adj[1]);
                bv2[e] = f2bf(vr[o + 32 + nn] + adj[2]);
                bv3[e] = f2bf(vr[o + 48 + nn] + adj[3]);
            }
        }
        o0 = wmma_bf16(ap, bv0, o0);
        o1 = wmma_bf16(ap, bv1, o1);
        o2 = wmma_bf16(ap, bv2, o2);
        o3 = wmma_bf16(ap, bv3, o3);
    }

    // finalize + store to branch (B,T,D)
    #pragma unroll
    for (int r = 0; r < 8; ++r) {
        const float inv = 1.0f / Lrow[r];
        const int t = q0 + half * 8 + r;
        float* orow = branch + ((size_t)b * TT + t) * DD + h * DHH;
        orow[0  + nn] = o0[r] * inv;
        orow[16 + nn] = o1[r] * inv;
        orow[32 + nn] = o2[r] * inv;
        orow[48 + nn] = o3[r] * inv;
    }
}

// ---------------------------------------------------------------------------
// Grouped temporal conv (k=5, causal) + grouped FF (up/gelu/down) +
// residual + per-head GroupNorm. One block per (b, t).
// ---------------------------------------------------------------------------
__global__ __launch_bounds__(256) void conv_ff_gn_kernel(
    const float* __restrict__ branch,
    const float* __restrict__ tconv_w, const float* __restrict__ tconv_b,
    const float* __restrict__ ffu_w,   const float* __restrict__ ffu_b,
    const float* __restrict__ ffd_w,   const float* __restrict__ ffd_b,
    const float* __restrict__ gn_w,    const float* __restrict__ gn_b,
    float* __restrict__ normed)
{
    const int b = blockIdx.x >> 10;
    const int t = blockIdx.x & 1023;
    const int tid = threadIdx.x;

    __shared__ float xrows[5][DD];
    __shared__ float xcf[DD];
    __shared__ float hbuf[2 * DD];

    #pragma unroll
    for (int k = 0; k < 5; ++k) {
        const int ts = t - 4 + k;
        for (int d = tid; d < DD; d += 256)
            xrows[k][d] = (ts >= 0) ? branch[((size_t)b * TT + ts) * DD + d] : 0.0f;
    }
    __syncthreads();

    // temporal grouped conv + residual
    for (int o = tid; o < DD; o += 256) {
        const int g = o >> 6;
        const float* w = tconv_w + (size_t)o * (DHH * 5);
        float acc = tconv_b[o];
        for (int i = 0; i < DHH; ++i) {
            const float* wi = w + i * 5;
            const int c = g * DHH + i;
            #pragma unroll
            for (int k = 0; k < 5; ++k) acc += wi[k] * xrows[k][c];
        }
        xcf[o] = xrows[4][o] + acc;
    }
    __syncthreads();

    // grouped FF up + GeLU
    for (int u = tid; u < 2 * DD; u += 256) {
        const int g = u >> 7;
        const float* w = ffu_w + (size_t)u * DHH;
        float acc = ffu_b[u];
        for (int i = 0; i < DHH; ++i) acc += w[i] * xcf[g * DHH + i];
        hbuf[u] = gelu_f(acc);
    }
    __syncthreads();

    // grouped FF down + residual; then per-head GroupNorm (16-lane groups)
    float rv[4];
    const int d0 = tid * 4;
    #pragma unroll
    for (int j = 0; j < 4; ++j) {
        const int d = d0 + j;
        const int g = d >> 6;
        const float* w = ffd_w + (size_t)d * (2 * DHH);
        float acc = ffd_b[d];
        for (int i = 0; i < 2 * DHH; ++i) acc += w[i] * hbuf[g * 2 * DHH + i];
        rv[j] = xrows[4][d] + acc;
    }
    float s = rv[0] + rv[1] + rv[2] + rv[3];
    float s2 = rv[0]*rv[0] + rv[1]*rv[1] + rv[2]*rv[2] + rv[3]*rv[3];
    #pragma unroll
    for (int off = 8; off >= 1; off >>= 1) { s += __shfl_xor(s, off); s2 += __shfl_xor(s2, off); }
    const float mu = s * (1.0f / 64.0f);
    const float var = s2 * (1.0f / 64.0f) - mu * mu;
    const float rstd = rsqrtf(var + 1e-5f);
    #pragma unroll
    for (int j = 0; j < 4; ++j) {
        const int d = d0 + j;
        normed[((size_t)b * TT + t) * DD + d] = (rv[j] - mu) * rstd * gn_w[d] + gn_b[d];
    }
}

// ---------------------------------------------------------------------------
// NMDA state update from last timestep of normed heads.
// ---------------------------------------------------------------------------
__global__ __launch_bounds__(256) void nmda_kernel(
    const float* __restrict__ normed, const float* __restrict__ nmda,
    const float* __restrict__ decay, float* __restrict__ out)
{
    const int i = blockIdx.x * 256 + threadIdx.x;   // B*NH*DH = 4096
    if (i >= BB * NHH * DHH) return;
    const int b = i >> 10;
    const int hd = i & 1023;
    const float tau = 1.0f / (1.0f + __expf(-decay[hd]));
    const float bl = normed[((size_t)b * TT + (TT - 1)) * DD + hd];
    out[i] = tau * nmda[i] + (1.0f - tau) * bl;
}

// ---------------------------------------------------------------------------
// Row LayerNorm (D = 1024), one block per row.
// ---------------------------------------------------------------------------
__global__ __launch_bounds__(256) void layernorm_kernel(
    const float* __restrict__ x, const float* __restrict__ w,
    const float* __restrict__ bb, float* __restrict__ y)
{
    const int row = blockIdx.x;
    const int tid = threadIdx.x;
    const float* xr = x + (size_t)row * DD;
    float v[4]; float s = 0.0f, s2 = 0.0f;
    #pragma unroll
    for (int j = 0; j < 4; ++j) { v[j] = xr[tid + j * 256]; s += v[j]; s2 += v[j] * v[j]; }
    #pragma unroll
    for (int off = 16; off >= 1; off >>= 1) { s += __shfl_xor(s, off); s2 += __shfl_xor(s2, off); }
    __shared__ float rs[8], rs2[8];
    const int wave = tid >> 5, lane = tid & 31;
    if (lane == 0) { rs[wave] = s; rs2[wave] = s2; }
    __syncthreads();
    if (tid == 0) {
        float a = 0.0f, c = 0.0f;
        for (int i = 0; i < 8; ++i) { a += rs[i]; c += rs2[i]; }
        rs[0] = a; rs2[0] = c;
    }
    __syncthreads();
    s = rs[0]; s2 = rs2[0];
    const float mu = s * (1.0f / DD);
    const float var = s2 * (1.0f / DD) - mu * mu;
    const float rstd = rsqrtf(var + 1e-5f);
    #pragma unroll
    for (int j = 0; j < 4; ++j) {
        const int d = tid + j * 256;
        y[(size_t)row * DD + d] = (v[j] - mu) * rstd * w[d] + bb[d];
    }
}

// ---------------------------------------------------------------------------
// Launch
// ---------------------------------------------------------------------------
extern "C" void kernel_launch(void* const* d_in, const int* in_sizes, int n_in,
                              void* d_out, int out_size, void* d_ws, size_t ws_size,
                              hipStream_t stream)
{
    const float* x          = (const float*)d_in[0];
    const float* nmda       = (const float*)d_in[1];
    const float* qkv_w      = (const float*)d_in[2];
    const float* nmda_decay = (const float*)d_in[3];
    const float* tconv_w    = (const float*)d_in[4];
    const float* tconv_b    = (const float*)d_in[5];
    const float* ffu_w      = (const float*)d_in[6];
    const float* ffu_b      = (const float*)d_in[7];
    const float* ffd_w      = (const float*)d_in[8];
    const float* ffd_b      = (const float*)d_in[9];
    const float* gn_w       = (const float*)d_in[10];
    const float* gn_b       = (const float*)d_in[11];
    const float* soma_w     = (const float*)d_in[12];
    const float* ln_w       = (const float*)d_in[13];
    const float* ln_b       = (const float*)d_in[14];
    const float* w1         = (const float*)d_in[15];
    const float* b1         = (const float*)d_in[16];
    const float* w2         = (const float*)d_in[17];
    const float* b2         = (const float*)d_in[18];

    const int M = BB * TT;                       // 4096
    float* ws      = (float*)d_ws;
    float* qkvbuf  = ws;                         // M * 3D (reused as MLP hidden)
    float* branch  = qkvbuf + (size_t)M * 3 * DD;
    float* normed  = branch + (size_t)M * DD;
    float* soma    = normed + (size_t)M * DD;
    float* lns     = soma   + (size_t)M * DD;
    float* out_soma = (float*)d_out;
    float* out_nmda = out_soma + (size_t)BB * TT * DD;

    dim3 blk(256);

    // 1) qkv = x @ qkv_w
    gemm_bf16_kernel<0><<<dim3(3 * DD / 128, M / 64), blk, 0, stream>>>(
        x, qkv_w, nullptr, nullptr, qkvbuf, M, 3 * DD, DD);
    // 2) causal attention (+NMDA bias on V) -> branch
    attn_kernel<<<BB * NHH * (TT / 16) / 8, blk, 0, stream>>>(
        qkvbuf, nmda, nmda_decay, branch);
    // 3) temporal conv + FF + GroupNorm -> normed
    conv_ff_gn_kernel<<<BB * TT, blk, 0, stream>>>(
        branch, tconv_w, tconv_b, ffu_w, ffu_b, ffd_w, ffd_b, gn_w, gn_b, normed);
    // 4) NMDA state update -> out
    nmda_kernel<<<(BB * NHH * DHH + 255) / 256, blk, 0, stream>>>(
        normed, nmda, nmda_decay, out_nmda);
    // 5) soma = normed @ soma_w
    gemm_bf16_kernel<0><<<dim3(DD / 128, M / 64), blk, 0, stream>>>(
        normed, soma_w, nullptr, nullptr, soma, M, DD, DD);
    // 6) lns = LayerNorm(soma)
    layernorm_kernel<<<M, blk, 0, stream>>>(soma, ln_w, ln_b, lns);
    // 7) h = gelu(lns @ w1 + b1)   (reuse qkv buffer)
    gemm_bf16_kernel<1><<<dim3(SOMA_HH / 128, M / 64), blk, 0, stream>>>(
        lns, w1, b1, nullptr, qkvbuf, M, SOMA_HH, DD);
    // 8) out = soma + (h @ w2 + b2)
    gemm_bf16_kernel<2><<<dim3(DD / 128, M / 64), blk, 0, stream>>>(
        qkvbuf, w2, b2, soma, out_soma, M, DD, SOMA_HH);
}
